// CausalSelfAttention_75788992905417
// MI455X (gfx1250) — matrix operand order repaired and checked by reading, not compiled, not run on hardware
//
#include <hip/hip_runtime.h>

// ---------------- CDNA5 WMMA / TDM types ----------------
typedef __attribute__((ext_vector_type(16))) __bf16 v16bf;
typedef __attribute__((ext_vector_type(8)))  float  v8f;
typedef __attribute__((ext_vector_type(4)))  unsigned int v4u;
typedef __attribute__((ext_vector_type(8)))  int v8i;
typedef __attribute__((ext_vector_type(4)))  int v4i;

union Frag {
    v16bf h;
    unsigned int u[8];
};

// Sizes (compile-time constants for this problem)
#define NB   4
#define SS   2048
#define EE   1024
#define HH   16
#define DD   64
#define M_ROWS (NB * SS)          // 8192
#define N_KQV  (3 * EE)           // 3072

// ---------------- fragment loaders (ISA 7.12.2 layouts) ----------------
// A-matrix 16x32 bf16: lane m = lane&15, half = lane>>4.
// VGPR j<4: K = 2j + 8*half + pair ; j>=4: K = 16 + 2(j-4) + 8*half + pair
__device__ __forceinline__ void load_a_frag(Frag& f, const __bf16* row, int half) {
#pragma unroll
    for (int j = 0; j < 8; ++j) {
        int kb = ((j < 4) ? (j * 2) : (16 + (j - 4) * 2)) + half * 8;
        f.u[j] = *(const unsigned int*)(row + kb);
    }
}

// B-matrix 32x16 bf16: lane n = lane&15, half = lane>>4.
// VGPR j<4: K = 2j + 8*half + pair ; j>=4: K = 16 + 2(j-4) + 8*half + pair.  `krow` points at B^T row n (contiguous along K).
__device__ __forceinline__ void load_b_frag(Frag& f, const __bf16* krow, int half) {
#pragma unroll
    for (int j = 0; j < 8; ++j) {
        int kb = ((j < 4) ? (j * 2) : (16 + (j - 4) * 2)) + half * 8;
        f.u[j] = *(const unsigned int*)(krow + kb);
    }
}

// ---------------- TDM: 2-D tile  global -> LDS  (D# per ISA ch.8) ----------------
// tensor element = 2 bytes (bf16). Loads tile_d1 rows of tile_d0 contiguous elements,
// row stride = stride0 elements, packed contiguously into LDS at lds_off.
__device__ __forceinline__ void tdm_load_2d(const __bf16* tile_base, unsigned lds_off,
                                            unsigned tensor_d0, unsigned tensor_d1,
                                            unsigned long long stride0,
                                            unsigned tile_d0, unsigned tile_d1) {
    unsigned long long ga = (unsigned long long)(size_t)tile_base;
    v4u g0;
    g0[0] = 1u;                                              // count=1, user descriptor
    g0[1] = lds_off;                                         // lds_addr (bytes)
    g0[2] = (unsigned)(ga & 0xffffffffu);                    // global_addr[31:0]
    g0[3] = (unsigned)((ga >> 32) & 0x01ffffffu) | (2u << 30); // global_addr[56:32], type=2
    v8i g1;
    g1[0] = (int)(1u << 16);                                 // wg_mask=0, data_size=1 (2B)
    g1[1] = (int)((tensor_d0 & 0xffffu) << 16);              // tensor_dim0[15:0]
    g1[2] = (int)(((tensor_d0 >> 16) & 0xffffu) | ((tensor_d1 & 0xffffu) << 16));
    g1[3] = (int)(((tensor_d1 >> 16) & 0xffffu) | ((tile_d0 & 0xffffu) << 16));
    g1[4] = (int)(tile_d1 & 0xffffu);                        // tile_dim1, tile_dim2=0
    g1[5] = (int)(stride0 & 0xffffffffu);                    // tensor_dim0_stride[31:0]
    g1[6] = (int)((stride0 >> 32) & 0xffffu);                // stride0[47:32], stride1 lo=0
    g1[7] = 0;
    v4i zz4;
    zz4[0] = 0; zz4[1] = 0; zz4[2] = 0; zz4[3] = 0;
    v8i zz8;
#pragma unroll
    for (int i = 0; i < 8; ++i) zz8[i] = 0;
    // 6-arg variant (clang-23 / therock-10.0 headers)
    __builtin_amdgcn_tensor_load_to_lds(g0, g1, zz4, zz4, zz8, 0);
}

// ---------------- fp32 -> bf16 convert ----------------
__global__ __launch_bounds__(256) void cvt_f32_bf16(const float* __restrict__ in,
                                                    __bf16* __restrict__ out, int n) {
    int i = blockIdx.x * 256 + threadIdx.x;
    if (i < n) out[i] = (__bf16)in[i];
}

// ---------------- GEMM1: kqv = x @ W_kqv^T + b, scatter to head-major K/Q/V^T ----------------
// one wave per 32(M) x 64(N) tile.  grid = (M/32, N/64)
__global__ __launch_bounds__(32) void gemm_kqv(const __bf16* __restrict__ xb,
                                               const __bf16* __restrict__ wb,
                                               const float*  __restrict__ bias,
                                               __bf16* __restrict__ kout,
                                               __bf16* __restrict__ qout,
                                               __bf16* __restrict__ vtout) {
    const int lane = threadIdx.x & 31;
    const int nloc = lane & 15;
    const int half = lane >> 4;
    const int m0 = blockIdx.x * 32;
    const int n0 = blockIdx.y * 64;

    v8f acc[2][4];
#pragma unroll
    for (int mi = 0; mi < 2; ++mi)
#pragma unroll
        for (int t = 0; t < 4; ++t)
#pragma unroll
            for (int r = 0; r < 8; ++r) acc[mi][t][r] = 0.0f;

    for (int k0 = 0; k0 < EE; k0 += 32) {
        Frag a[2];
#pragma unroll
        for (int mi = 0; mi < 2; ++mi)
            load_a_frag(a[mi], xb + (size_t)(m0 + mi * 16 + nloc) * EE + k0, half);
#pragma unroll
        for (int t = 0; t < 4; ++t) {
            Frag b;
            load_b_frag(b, wb + (size_t)(n0 + t * 16 + nloc) * EE + k0, half);
#pragma unroll
            for (int mi = 0; mi < 2; ++mi)
                acc[mi][t] = __builtin_amdgcn_wmma_f32_16x16x32_bf16(
                    false, a[mi].h, false, b.h, (short)0, acc[mi][t], false, false);
        }
    }

#pragma unroll
    for (int t = 0; t < 4; ++t) {
        const int nglob = n0 + t * 16 + nloc;
        const float bv = bias[nglob];
        const int tensor = nglob >> 10;       // 0=k, 1=q, 2=v
        const int e = nglob & (EE - 1);
        const int h = e >> 6;
        const int d = e & 63;
#pragma unroll
        for (int mi = 0; mi < 2; ++mi)
#pragma unroll
            for (int r = 0; r < 8; ++r) {
                const int row = m0 + mi * 16 + r + half * 8;   // global token row
                const int b = row >> 11;
                const int s = row & (SS - 1);
                const __bf16 hv = (__bf16)(acc[mi][t][r] + bv);
                const int bh = b * HH + h;
                if (tensor == 0)
                    kout[((size_t)bh * SS + s) * DD + d] = hv;
                else if (tensor == 1)
                    qout[((size_t)bh * SS + s) * DD + d] = hv;
                else
                    vtout[((size_t)bh * DD + d) * SS + s] = hv;   // V^T [bh][d][s]
            }
    }
}

// ---------------- block-cooperative flash attention with TDM double buffering ----------
// block = 128 threads (4 waves) handles one (b,h) and 4 consecutive 16-query tiles.
// K/V 32-key tiles staged once per block into LDS by the Tensor Data Mover.
__global__ __launch_bounds__(128) void attn_fwd(const __bf16* __restrict__ qbuf,
                                                const __bf16* __restrict__ kbuf,
                                                const __bf16* __restrict__ vtbuf,
                                                __bf16* __restrict__ ybuf) {
    __shared__ __bf16 sK[2][32 * DD];    // [key][d]
    __shared__ __bf16 sVT[2][DD * 32];   // [d][key]
    __shared__ __bf16 sP[4][16 * 32];

    const int wid  = threadIdx.x >> 5;
    const int lane = threadIdx.x & 31;
    const int nloc = lane & 15;
    const int half = lane >> 4;

    const int bh  = blockIdx.x >> 5;               // uniform (SGPR): 0..63
    const int qt0 = (blockIdx.x & 31) * 4;         // first of 4 query tiles
    const int q0 = (qt0 + wid) * 16;               // this wave's query tile
    const int my_kend  = q0 + 16;
    const int blk_kend = qt0 * 16 + 64;            // max keys needed in block

    const __bf16* Q  = qbuf  + (size_t)bh * SS * DD;
    const __bf16* K  = kbuf  + (size_t)bh * SS * DD;
    const __bf16* VT = vtbuf + (size_t)bh * DD * SS;

    // Q fragments for the two 32-wide d-chunks (loaded once, from global)
    Frag aq[2];
#pragma unroll
    for (int c = 0; c < 2; ++c)
        load_a_frag(aq[c], Q + (size_t)(q0 + nloc) * DD + c * 32, half);

    float mrow[8], lrow[8];
    v8f acc[4];
#pragma unroll
    for (int r = 0; r < 8; ++r) { mrow[r] = -1.0e30f; lrow[r] = 0.0f; }
#pragma unroll
    for (int t = 0; t < 4; ++t)
#pragma unroll
        for (int r = 0; r < 8; ++r) acc[t][r] = 0.0f;

    const float scale = 0.125f;                    // 1/sqrt(64)

    // prologue: wave0 DMAs tile kb=0 into buffer 0
    if (wid == 0) {
        tdm_load_2d(K, (unsigned)(size_t)&sK[0][0], DD, 32, DD, DD, 32);
        tdm_load_2d(VT, (unsigned)(size_t)&sVT[0][0], SS, DD, SS, 32, DD);
    }

    int cur = 0;
    for (int kb = 0; kb < blk_kend; kb += 32) {
        if (wid == 0) {
            if (kb + 32 < blk_kend) {   // prefetch next tile into other buffer
                tdm_load_2d(K + (size_t)(kb + 32) * DD,
                            (unsigned)(size_t)&sK[cur ^ 1][0], DD, 32, DD, DD, 32);
                tdm_load_2d(VT + (kb + 32),
                            (unsigned)(size_t)&sVT[cur ^ 1][0], SS, DD, SS, 32, DD);
                __builtin_amdgcn_s_wait_tensorcnt(2);  // current tile's 2 ops done
            } else {
                __builtin_amdgcn_s_wait_tensorcnt(0);
            }
        }
        __syncthreads();

        if (kb < my_kend) {    // wave-uniform: EXEC stays all-ones for WMMA
            const __bf16* kt = sK[cur];
            const __bf16* vt = sVT[cur];

            // ---- scores: two 16-key subtiles, contraction over d (two chunks) ----
            v8f s[2];
#pragma unroll
            for (int st = 0; st < 2; ++st) {
                v8f c;
#pragma unroll
                for (int r = 0; r < 8; ++r) c[r] = 0.0f;
                const __bf16* krow = kt + (st * 16 + nloc) * DD;
#pragma unroll
                for (int c2 = 0; c2 < 2; ++c2) {
                    Frag bk;
                    load_b_frag(bk, krow + c2 * 32, half);
                    c = __builtin_amdgcn_wmma_f32_16x16x32_bf16(
                        false, aq[c2].h, false, bk.h, (short)0, c, false, false);
                }
                s[st] = c;
            }

            // ---- scale + causal mask + row max ----
            float rmax[8];
#pragma unroll
            for (int r = 0; r < 8; ++r) {
                const int qg = q0 + r + half * 8;
                float v0 = s[0][r] * scale; if (kb + nloc > qg)      v0 = -3.0e38f;
                float v1 = s[1][r] * scale; if (kb + 16 + nloc > qg) v1 = -3.0e38f;
                s[0][r] = v0; s[1][r] = v1;
                rmax[r] = fmaxf(v0, v1);
            }
#pragma unroll
            for (int mk = 1; mk < 16; mk <<= 1)
#pragma unroll
                for (int r = 0; r < 8; ++r)
                    rmax[r] = fmaxf(rmax[r], __shfl_xor(rmax[r], mk, 32));

            // ---- online softmax update ----
            float fac[8], psum[8];
#pragma unroll
            for (int r = 0; r < 8; ++r) {
                const float mn = fmaxf(mrow[r], rmax[r]);
                fac[r]  = __expf(mrow[r] - mn);
                mrow[r] = mn;
                const float p0 = __expf(s[0][r] - mn);
                const float p1 = __expf(s[1][r] - mn);
                s[0][r] = p0; s[1][r] = p1;
                psum[r] = p0 + p1;
            }
#pragma unroll
            for (int mk = 1; mk < 16; mk <<= 1)
#pragma unroll
                for (int r = 0; r < 8; ++r)
                    psum[r] += __shfl_xor(psum[r], mk, 32);
#pragma unroll
            for (int r = 0; r < 8; ++r) {
                lrow[r] = lrow[r] * fac[r] + psum[r];
#pragma unroll
                for (int t = 0; t < 4; ++t) acc[t][r] *= fac[r];
            }

            // ---- re-layout P via per-wave LDS (C-layout -> A-fragment) ----
            __bf16* sp = sP[wid];
#pragma unroll
            for (int r = 0; r < 8; ++r) {
                const int m = r + half * 8;
                sp[m * 32 + nloc]      = (__bf16)s[0][r];
                sp[m * 32 + 16 + nloc] = (__bf16)s[1][r];
            }
            Frag ap;
            load_a_frag(ap, sp + nloc * 32, half);

            // ---- acc += P @ V ----
#pragma unroll
            for (int t = 0; t < 4; ++t) {
                Frag bv;
                load_b_frag(bv, vt + (t * 16 + nloc) * 32, half);
                acc[t] = __builtin_amdgcn_wmma_f32_16x16x32_bf16(
                    false, ap.h, false, bv.h, (short)0, acc[t], false, false);
            }
        }
        __syncthreads();
        cur ^= 1;
    }

    // ---- normalize + write y (bf16, [B,S,E] row-major) ----
    const int h = bh & (HH - 1);
    const int b = bh >> 4;
#pragma unroll
    for (int t = 0; t < 4; ++t)
#pragma unroll
        for (int r = 0; r < 8; ++r) {
            const int m = r + half * 8;
            const float o = acc[t][r] / lrow[r];
            ybuf[(size_t)(b * SS + q0 + m) * EE + h * DD + t * 16 + nloc] = (__bf16)o;
        }
}

// ---------------- GEMM2: out = y @ W_proj^T + b (fp32 out) ----------------
__global__ __launch_bounds__(32) void gemm_proj(const __bf16* __restrict__ yb,
                                                const __bf16* __restrict__ wb,
                                                const float*  __restrict__ bias,
                                                float* __restrict__ out) {
    const int lane = threadIdx.x & 31;
    const int nloc = lane & 15;
    const int half = lane >> 4;
    const int m0 = blockIdx.x * 32;
    const int n0 = blockIdx.y * 64;

    v8f acc[2][4];
#pragma unroll
    for (int mi = 0; mi < 2; ++mi)
#pragma unroll
        for (int t = 0; t < 4; ++t)
#pragma unroll
            for (int r = 0; r < 8; ++r) acc[mi][t][r] = 0.0f;

    for (int k0 = 0; k0 < EE; k0 += 32) {
        Frag a[2];
#pragma unroll
        for (int mi = 0; mi < 2; ++mi)
            load_a_frag(a[mi], yb + (size_t)(m0 + mi * 16 + nloc) * EE + k0, half);
#pragma unroll
        for (int t = 0; t < 4; ++t) {
            Frag b;
            load_b_frag(b, wb + (size_t)(n0 + t * 16 + nloc) * EE + k0, half);
#pragma unroll
            for (int mi = 0; mi < 2; ++mi)
                acc[mi][t] = __builtin_amdgcn_wmma_f32_16x16x32_bf16(
                    false, a[mi].h, false, b.h, (short)0, acc[mi][t], false, false);
        }
    }

#pragma unroll
    for (int t = 0; t < 4; ++t) {
        const int nglob = n0 + t * 16 + nloc;
        const float bv = bias[nglob];
#pragma unroll
        for (int mi = 0; mi < 2; ++mi)
#pragma unroll
            for (int r = 0; r < 8; ++r) {
                const int row = m0 + mi * 16 + r + half * 8;
                out[(size_t)row * EE + nglob] = acc[mi][t][r] + bv;
            }
    }
}

// ---------------- host launcher ----------------
extern "C" void kernel_launch(void* const* d_in, const int* in_sizes, int n_in,
                              void* d_out, int out_size, void* d_ws, size_t ws_size,
                              hipStream_t stream) {
    const float* x     = (const float*)d_in[0];
    const float* Wkqv  = (const float*)d_in[1];
    const float* bkqv  = (const float*)d_in[2];
    const float* Wproj = (const float*)d_in[3];
    const float* bproj = (const float*)d_in[4];
    float* out = (float*)d_out;

    char* ws = (char*)d_ws;
    size_t off = 0;
    __bf16* xb     = (__bf16*)(ws + off); off += (size_t)M_ROWS * EE * 2;   // 16 MB
    __bf16* wkqvb  = (__bf16*)(ws + off); off += (size_t)N_KQV * EE * 2;    // 6 MB
    __bf16* wprojb = (__bf16*)(ws + off); off += (size_t)EE * EE * 2;       // 2 MB
    __bf16* kbuf   = (__bf16*)(ws + off); off += (size_t)M_ROWS * EE * 2;   // 16 MB
    __bf16* qbuf   = (__bf16*)(ws + off); off += (size_t)M_ROWS * EE * 2;   // 16 MB
    __bf16* vtbuf  = (__bf16*)(ws + off); off += (size_t)M_ROWS * EE * 2;   // 16 MB
    __bf16* ybuf   = (__bf16*)(ws + off); off += (size_t)M_ROWS * EE * 2;   // 16 MB

    const int nX  = M_ROWS * EE;     // 8388608
    const int nW1 = N_KQV * EE;      // 3145728
    const int nW2 = EE * EE;         // 1048576
    cvt_f32_bf16<<<nX / 256, 256, 0, stream>>>(x, xb, nX);
    cvt_f32_bf16<<<nW1 / 256, 256, 0, stream>>>(Wkqv, wkqvb, nW1);
    cvt_f32_bf16<<<nW2 / 256, 256, 0, stream>>>(Wproj, wprojb, nW2);

    dim3 g1(M_ROWS / 32, N_KQV / 64);       // 256 x 48
    gemm_kqv<<<g1, 32, 0, stream>>>(xb, wkqvb, bkqv, kbuf, qbuf, vtbuf);

    attn_fwd<<<NB * HH * (SS / 64), 128, 0, stream>>>(qbuf, kbuf, vtbuf, ybuf);

    dim3 g2(M_ROWS / 32, EE / 64);          // 256 x 16
    gemm_proj<<<g2, 32, 0, stream>>>(ybuf, wprojb, bproj, out);
}
